// HydraBlock_6064493822037
// MI455X (gfx1250) — compile-verified, matrix operand order checked by reading, not run on hardware
//
#include <hip/hip_runtime.h>
#include <hip/hip_bf16.h>
#include <math.h>

// ---------------------------------------------------------------------------
// CDNA5 WMMA types (wave32): v_wmma_f32_16x16x32_bf16
// ---------------------------------------------------------------------------
typedef __attribute__((ext_vector_type(16))) __bf16 v16bf;
typedef __attribute__((ext_vector_type(8)))  __bf16 v8bf;
typedef __attribute__((ext_vector_type(8)))  float  v8f;

union FragU { v16bf v; v8bf h[2]; };

__device__ __forceinline__ v8f wmma_bf16(v16bf a, v16bf b, v8f c) {
  return __builtin_amdgcn_wmma_f32_16x16x32_bf16(
      false, a, false, b, (short)0, c, false, false);
}

// 16-bit A/B fragment (ISA 7.12.2): lane L -> row/col = L%16, half = L/16.
// Elements 0..7 <-> k = 8*half + j ; elements 8..15 <-> k = 16 + 8*half + j.
// Two contiguous 16B runs -> two b128 loads (global or LDS).
__device__ __forceinline__ v16bf ld_frag(const __bf16* row, int half) {
  FragU u;
  u.h[0] = *(const v8bf*)(row + 8 * half);
  u.h[1] = *(const v8bf*)(row + 16 + 8 * half);
  return u.v;
}

// ---------------------------------------------------------------------------
// GEMM: C[M,N] (+)= A[M,K] * Bt[N,K]^T (+bias).  A,Bt bf16; C f32 or bf16.
// 128 threads = 4 waves; wave w owns a 64 x (16*NSUB) tile: rows
// [bm+64w, +64), cols [bn, bn+16*NSUB).  Fragments load straight from global;
// k-loop software-pipelined (double buffered): 16 WMMAs cover the latency of
// the next step's 16 b128 loads (1.0 loads/WMMA).
// Requires M%256==0, K%32==0, N%(16*NSUB)==0.
// ---------------------------------------------------------------------------
template <int NSUB>
__global__ __launch_bounds__(128) void gemm_bf16v4_kernel(
    const __bf16* __restrict__ A, const __bf16* __restrict__ Bt,
    const float* __restrict__ bias, float* __restrict__ Cf,
    __bf16* __restrict__ Cbf, int M, int N, int K,
    int accum, int writebf)
{
  const int bm = blockIdx.y * 256;
  const int bn = blockIdx.x * (16 * NSUB);
  const int wave = threadIdx.x >> 5, lane = threadIdx.x & 31;
  const int half = lane >> 4, l16 = lane & 15;
  const int wm = bm + wave * 64;

  v8f acc[4][NSUB] = {};
  const __bf16* arow[4];
#pragma unroll
  for (int i = 0; i < 4; ++i) arow[i] = A + (size_t)(wm + i * 16 + l16) * K;
  const __bf16* brow[NSUB];
#pragma unroll
  for (int j = 0; j < NSUB; ++j) brow[j] = Bt + (size_t)(bn + j * 16 + l16) * K;

  // Prologue: fragments for k0 = 0.
  v16bf af[4], bf[NSUB];
#pragma unroll
  for (int i = 0; i < 4; ++i) af[i] = ld_frag(arow[i], half);
#pragma unroll
  for (int j = 0; j < NSUB; ++j) bf[j] = ld_frag(brow[j], half);

  for (int k0 = 32; k0 < K; k0 += 32) {
    // Issue next step's loads first...
    v16bf na[4], nb[NSUB];
#pragma unroll
    for (int i = 0; i < 4; ++i) na[i] = ld_frag(arow[i] + k0, half);
#pragma unroll
    for (int j = 0; j < NSUB; ++j) nb[j] = ld_frag(brow[j] + k0, half);
    // ...then compute on the previous step's fragments.
#pragma unroll
    for (int j = 0; j < NSUB; ++j)
#pragma unroll
      for (int i = 0; i < 4; ++i)
        acc[i][j] = wmma_bf16(af[i], bf[j], acc[i][j]);
#pragma unroll
    for (int i = 0; i < 4; ++i) af[i] = na[i];
#pragma unroll
    for (int j = 0; j < NSUB; ++j) bf[j] = nb[j];
  }
#pragma unroll
  for (int j = 0; j < NSUB; ++j)
#pragma unroll
    for (int i = 0; i < 4; ++i)
      acc[i][j] = wmma_bf16(af[i], bf[j], acc[i][j]);

  // C/D layout: lane -> col = l16 (within 16-tile), rows = 8*half + r
#pragma unroll
  for (int j = 0; j < NSUB; ++j) {
    const int col = bn + j * 16 + l16;
    const float bv = bias ? bias[col] : 0.f;
#pragma unroll
    for (int i = 0; i < 4; ++i) {
#pragma unroll
      for (int r = 0; r < 8; ++r) {
        const int row = wm + i * 16 + 8 * half + r;
        const size_t idx = (size_t)row * N + col;
        float v = acc[i][j][r] + bv;
        if (accum) v += Cf[idx];
        if (writebf) Cbf[idx] = (__bf16)v;
        else         Cf[idx] = v;
      }
    }
  }
}

// ---------------------------------------------------------------------------
// Flash attention (causal), dh=64, 8 heads, bf16 in/out.
// 128 threads = 4 waves = 64 queries of one (b,h). Key blocks of 32.
// Q/K fragments direct from global; V transposed via LDS; P via LDS.
// ---------------------------------------------------------------------------
#define FA_T 2048
#define FA_D 512

__global__ __launch_bounds__(128) void mla_flash_attn_kernel(
    const __bf16* __restrict__ Q, const __bf16* __restrict__ Kb,
    const __bf16* __restrict__ Vb, __bf16* __restrict__ O)
{
  const int qt = blockIdx.x, h = blockIdx.y, b = blockIdx.z;
  const int tid = threadIdx.x;
  const int wave = tid >> 5, lane = tid & 31;
  const int half = lane >> 4, l16 = lane & 15;
  const int q0 = qt * 64 + wave * 16;
  const size_t hoff = (size_t)h * 64;
  const size_t bbase = (size_t)b * FA_T;

  __shared__ __align__(16) __bf16 sV[64][32];      // dh-major (transposed)
  __shared__ __align__(16) __bf16 sP[4][16][32];   // per-wave P tile

  v16bf qf[2];
  {
    const __bf16* qr = Q + (bbase + q0 + l16) * FA_D + hoff;
#pragma unroll
    for (int c = 0; c < 2; ++c) qf[c] = ld_frag(qr + c * 32, half);
  }

  v8f oacc[4] = {};
  float m_run[8], l_run[8];
#pragma unroll
  for (int r = 0; r < 8; ++r) { m_run[r] = -1e30f; l_run[r] = 0.f; }

  const int kmax = qt * 64 + 64;
  for (int j0 = 0; j0 < kmax; j0 += 32) {
    __syncthreads();                                // prior iter's sV reads done
    // Stage V block transposed: 32 keys x 64 dh -> sV[dh][key]
#pragma unroll
    for (int rep = 0; rep < 2; ++rep) {
      int linear = tid + rep * 128;                 // 0..255
      int kk = linear & 31, c8 = (linear >> 5) * 8;
      v8bf vv = *(const v8bf*)(Vb + (bbase + j0 + kk) * FA_D + hoff + c8);
#pragma unroll
      for (int j = 0; j < 8; ++j) sV[c8 + j][kk] = vv[j];
    }
    __syncthreads();

    // S = Q K^T (16 q x 32 keys), K fragments straight from global bf16.
    v8f s0 = {}, s1 = {};
    const __bf16* K0 = Kb + (bbase + j0 + l16) * FA_D + hoff;
    const __bf16* K1 = K0 + (size_t)16 * FA_D;
#pragma unroll
    for (int c = 0; c < 2; ++c) {
      s0 = wmma_bf16(qf[c], ld_frag(K0 + c * 32, half), s0);
      s1 = wmma_bf16(qf[c], ld_frag(K1 + c * 32, half), s1);
    }

    // Online softmax (lane: key col = l16 / 16+l16, rows = 8*half + r)
    const float scale = 0.125f;
    float pscale[8];
#pragma unroll
    for (int r = 0; r < 8; ++r) {
      int qrow = q0 + 8 * half + r;
      float a  = (j0 + l16 <= qrow)      ? s0[r] * scale : -1e30f;
      float c2 = (j0 + 16 + l16 <= qrow) ? s1[r] * scale : -1e30f;
      float mx = fmaxf(a, c2);
      for (int off = 1; off < 16; off <<= 1) mx = fmaxf(mx, __shfl_xor(mx, off, 32));
      float mnew = fmaxf(m_run[r], mx);
      float p0 = (a  <= -1e29f) ? 0.f : expf(a  - mnew);
      float p1 = (c2 <= -1e29f) ? 0.f : expf(c2 - mnew);
      s0[r] = p0; s1[r] = p1;
      float rs = p0 + p1;
      for (int off = 1; off < 16; off <<= 1) rs += __shfl_xor(rs, off, 32);
      float corr = (m_run[r] <= -1e29f) ? 0.f : expf(m_run[r] - mnew);
      l_run[r] = l_run[r] * corr + rs;
      m_run[r] = mnew;
      pscale[r] = corr;
    }
#pragma unroll
    for (int c = 0; c < 4; ++c)
#pragma unroll
      for (int r = 0; r < 8; ++r) oacc[c][r] *= pscale[r];

    // P: C-layout -> A-fragment layout through per-wave LDS region.
#pragma unroll
    for (int r = 0; r < 8; ++r) {
      sP[wave][8 * half + r][l16]      = (__bf16)s0[r];
      sP[wave][8 * half + r][16 + l16] = (__bf16)s1[r];
    }
    v16bf pf = ld_frag(&sP[wave][l16][0], half);    // same-wave DS ordering
#pragma unroll
    for (int c = 0; c < 4; ++c) {
      v16bf vf = ld_frag(&sV[c * 16 + l16][0], half);
      oacc[c] = wmma_bf16(pf, vf, oacc[c]);
    }
  }

#pragma unroll
  for (int c = 0; c < 4; ++c)
#pragma unroll
    for (int r = 0; r < 8; ++r) {
      int qrow = q0 + 8 * half + r;
      int dcol = c * 16 + l16;
      O[(bbase + qrow) * FA_D + hoff + dcol] = (__bf16)(oacc[c][r] / l_run[r]);
    }
}

// ---------------------------------------------------------------------------
// Weight transpose + f32->bf16:  Wt[n*K+k] = (bf16)W[k*N+n]
// ---------------------------------------------------------------------------
__global__ __launch_bounds__(256) void wtrans_kernel(const float* __restrict__ W,
                                                     __bf16* __restrict__ Wt,
                                                     int Kd, int Nd) {
  int i = blockIdx.x * 256 + threadIdx.x;
  if (i >= Kd * Nd) return;
  int n = i / Kd, k = i - n * Kd;
  Wt[i] = (__bf16)W[(size_t)k * Nd + n];
}

// ---------------------------------------------------------------------------
// Small kernels
// ---------------------------------------------------------------------------
#define NBLK 5
#define BS 409
#define TL (NBLK * BS)   // 2045

__global__ __launch_bounds__(256) void blk_mean_kernel(const float* __restrict__ x,
                                                       float* __restrict__ bm) {
  int bi = blockIdx.x;
  int b = bi / NBLK, nb = bi % NBLK;
  const float* base = x + ((size_t)b * 2048 + nb * BS) * 512;
  int d = threadIdx.x;
  float s0 = 0.f, s1 = 0.f;
  for (int r = 0; r < BS; ++r) {
    s0 += base[(size_t)r * 512 + d];
    s1 += base[(size_t)r * 512 + d + 256];
  }
  bm[(size_t)bi * 512 + d] = s0 / (float)BS;
  bm[(size_t)bi * 512 + d + 256] = s1 / (float)BS;
}

__global__ __launch_bounds__(32) void tg_gate_kernel(const float* __restrict__ bm,
                                                     const float* __restrict__ tgw,
                                                     const float* __restrict__ tgb,
                                                     float* __restrict__ imp) {
  int b = blockIdx.x, t = threadIdx.x;
  __shared__ float lg[NBLK][NBLK];
  if (t < NBLK * NBLK) {
    int i = t / NBLK, j = t % NBLK;
    float s = tgb[j];
    const float* row = bm + ((size_t)b * NBLK + i) * 512;
    for (int d = 0; d < 512; ++d) s += row[d] * tgw[d * NBLK + j];
    lg[i][j] = s;
  }
  __syncthreads();
  if (t < NBLK) {
    float mx = -1e30f;
    for (int j = 0; j < NBLK; ++j) mx = fmaxf(mx, lg[t][j]);
    float sum = 0.f, ee[NBLK];
    for (int j = 0; j < NBLK; ++j) { ee[j] = expf(lg[t][j] - mx); sum += ee[j]; }
    float m = 0.f;
    for (int j = 0; j < NBLK; ++j) m += ee[j] / sum;
    imp[b * NBLK + t] = m / (float)NBLK;
  }
}

__global__ __launch_bounds__(256) void tg_scale_kernel(const float* __restrict__ x,
                                                       const float* __restrict__ imp,
                                                       float* __restrict__ xf,
                                                       __bf16* __restrict__ xfb, int total) {
  int i = blockIdx.x * 256 + threadIdx.x;
  if (i >= total) return;
  int t = (i >> 9) & 2047;
  int b = i >> 20;
  float v = 0.f;
  if (t < TL) v = x[i] * imp[b * NBLK + t / BS];
  xf[i] = v;
  xfb[i] = (__bf16)v;
}

__global__ __launch_bounds__(256) void pool15_kernel(const float* __restrict__ xf,
                                                     __bf16* __restrict__ mac, int total) {
  int i = blockIdx.x * 256 + threadIdx.x;
  if (i >= total) return;
  int d = i & 511;
  int t = (i >> 9) & 2047;
  size_t base = (size_t)(i >> 20) * 2048 * 512;
  int lo = t - 7 < 0 ? 0 : t - 7;
  int hi = t + 7 > 2047 ? 2047 : t + 7;
  float s = 0.f;
  for (int k = lo; k <= hi; ++k) s += xf[base + (size_t)k * 512 + d];
  mac[i] = (__bf16)(s / (float)(hi - lo + 1));
}

__global__ __launch_bounds__(256) void resid_rms_kernel(const float* __restrict__ a,
                                                        const float* __restrict__ b,
                                                        const float* __restrict__ scale,
                                                        float* __restrict__ out,
                                                        __bf16* __restrict__ obf) {
  int row = blockIdx.x, t = threadIdx.x;
  __shared__ float red[256];
  size_t base = (size_t)row * 512;
  float v0 = a[base + t] + b[base + t];
  float v1 = a[base + t + 256] + b[base + t + 256];
  red[t] = v0 * v0 + v1 * v1;
  __syncthreads();
  for (int s = 128; s > 0; s >>= 1) { if (t < s) red[t] += red[t + s]; __syncthreads(); }
  float inv = rsqrtf(red[0] / 512.f + 1e-6f);
  float o0 = scale[t] * v0 * inv;
  float o1 = scale[t + 256] * v1 * inv;
  out[base + t] = o0;
  out[base + t + 256] = o1;
  if (obf) { obf[base + t] = (__bf16)o0; obf[base + t + 256] = (__bf16)o1; }
}

__global__ __launch_bounds__(256) void router_kernel(const float* __restrict__ x1,
                                                     const float* __restrict__ wr,
                                                     const float* __restrict__ br,
                                                     float* __restrict__ rw, int rows) {
  int row = blockIdx.x * 256 + threadIdx.x;
  if (row >= rows) return;
  const float* xr = x1 + (size_t)row * 512;
  float l[4];
  for (int e = 0; e < 4; ++e) l[e] = br[e];
  for (int d = 0; d < 512; ++d) {
    float xv = xr[d];
    for (int e = 0; e < 4; ++e) l[e] += xv * wr[d * 4 + e];
  }
  float mx = fmaxf(fmaxf(l[0], l[1]), fmaxf(l[2], l[3]));
  float s = 0.f;
  for (int e = 0; e < 4; ++e) { l[e] = expf(l[e] - mx); s += l[e]; }
  for (int e = 0; e < 4; ++e) rw[(size_t)row * 4 + e] = l[e] / s;
}

__global__ __launch_bounds__(256) void sigmul_kernel(const float* __restrict__ x1,
                                                     const float* __restrict__ g,
                                                     __bf16* __restrict__ xg, int total) {
  int i = blockIdx.x * 256 + threadIdx.x;
  if (i >= total) return;
  xg[i] = (__bf16)(x1[i] * (1.f / (1.f + expf(-g[i]))));
}

__global__ __launch_bounds__(256) void gelu_kernel(const float* __restrict__ h,
                                                   __bf16* __restrict__ hb, int total) {
  int i = blockIdx.x * 256 + threadIdx.x;
  if (i >= total) return;
  float x = h[i];
  hb[i] = (__bf16)(0.5f * x * (1.f + erff(x * 0.70710678118654752f)));
}

__global__ __launch_bounds__(256) void moe_comb_kernel(float* __restrict__ acc,
                                                       const float* __restrict__ eo,
                                                       const float* __restrict__ rw,
                                                       int e, int first, int total) {
  int i = blockIdx.x * 256 + threadIdx.x;
  if (i >= total) return;
  int row = i >> 9;
  float v = rw[(size_t)row * 4 + e] * eo[i];
  acc[i] = first ? v : acc[i] + v;
}

// ---------------------------------------------------------------------------
// Host orchestration
// ---------------------------------------------------------------------------
static inline void gemm(const __bf16* A, const __bf16* Bt, const float* bias,
                        float* Cf, __bf16* Cbf, int M, int N, int K,
                        int accum, int writebf, hipStream_t stream) {
  if (N % 64 == 0) {
    dim3 g(N / 64, M / 256);
    gemm_bf16v4_kernel<4><<<g, 128, 0, stream>>>(A, Bt, bias, Cf, Cbf, M, N, K,
                                                 accum, writebf);
  } else {  // N == 32
    dim3 g(N / 32, M / 256);
    gemm_bf16v4_kernel<2><<<g, 128, 0, stream>>>(A, Bt, bias, Cf, Cbf, M, N, K,
                                                 accum, writebf);
  }
}

static inline void wtrans(const float* W, __bf16* Wt, int Kd, int Nd, hipStream_t s) {
  int total = Kd * Nd;
  wtrans_kernel<<<(total + 255) / 256, 256, 0, s>>>(W, Wt, Kd, Nd);
}

extern "C" void kernel_launch(void* const* d_in, const int* in_sizes, int n_in,
                              void* d_out, int out_size, void* d_ws, size_t ws_size,
                              hipStream_t stream) {
  (void)in_sizes; (void)n_in; (void)out_size; (void)ws_size;

  const float* x    = (const float*)d_in[0];
  const float* tg_w = (const float*)d_in[1];
  const float* tg_b = (const float*)d_in[2];
  const float* mp_w = (const float*)d_in[3];
  const float* mp_b = (const float*)d_in[4];
  const float* fu_w = (const float*)d_in[5];
  const float* fu_b = (const float*)d_in[6];
  const float* wc   = (const float*)d_in[7];
  const float* wk   = (const float*)d_in[8];
  const float* wv   = (const float*)d_in[9];
  const float* wq   = (const float*)d_in[10];
  const float* wo   = (const float*)d_in[11];
  const float* s1   = (const float*)d_in[12];
  const float* s2   = (const float*)d_in[13];
  const float* wr   = (const float*)d_in[14];
  const float* br   = (const float*)d_in[15];
  const float* wg   = (const float*)d_in[16];
  const float* bg   = (const float*)d_in[17];
  const float* w1   = (const float*)d_in[18];
  const float* b1   = (const float*)d_in[19];
  const float* w2   = (const float*)d_in[20];
  const float* b2   = (const float*)d_in[21];
  float* out = (float*)d_out;

  const int Bb = 4, T = 2048, D = 512, Hd = 1024, NH = 8;
  const int BT = Bb * T;                      // 8192
  const size_t SZ = (size_t)BT * D;           // 4,194,304 elements

  // ---- workspace carve-up (256B aligned) ----
  char* p = (char*)d_ws;
  auto carve = [&](size_t bytes) { char* r = p; p += (bytes + 255) & ~(size_t)255; return r; };

  float* f1 = (float*)carve(SZ * 4);          // xf32 -> moe_acc
  float* f2 = (float*)carve(SZ * 4);          // xfu32 -> eo32
  float* f3 = (float*)carve(SZ * 4);          // attnproj32 -> gate32
  float* f4 = (float*)carve(SZ * 4);          // x1_32
  float* fh = (float*)carve((size_t)BT * Hd * 4);  // h32
  float* bm  = (float*)carve((size_t)Bb * NBLK * D * 4);
  float* imp = (float*)carve((size_t)Bb * NBLK * 4);
  float* rw  = (float*)carve((size_t)BT * 4 * 4);

  __bf16* g1 = (__bf16*)carve(SZ * 2);        // xf_bf  -> Qbf
  __bf16* g2 = (__bf16*)carve(SZ * 2);        // macro  -> Kbf
  __bf16* g3 = (__bf16*)carve(SZ * 2);        // mf_bf  -> Vbf
  __bf16* g4 = (__bf16*)carve(SZ * 2);        // xfu_bf -> attnout_bf
  __bf16* g5 = (__bf16*)carve(SZ * 2);        // x1_bf  -> xg_bf
  __bf16* gl = (__bf16*)carve((size_t)BT * 32 * 2);   // latent_bf
  __bf16* gh = (__bf16*)carve((size_t)BT * Hd * 2);   // h_bf

  __bf16* mp_t  = (__bf16*)carve((size_t)512 * 512 * 2);
  __bf16* fuT_t = (__bf16*)carve((size_t)512 * 512 * 2);
  __bf16* fuB_t = (__bf16*)carve((size_t)512 * 512 * 2);
  __bf16* wc_t  = (__bf16*)carve((size_t)32 * 512 * 2);
  __bf16* wk_t  = (__bf16*)carve((size_t)512 * 32 * 2);
  __bf16* wv_t  = (__bf16*)carve((size_t)512 * 32 * 2);
  __bf16* wq_t  = (__bf16*)carve((size_t)512 * 512 * 2);
  __bf16* wo_t  = (__bf16*)carve((size_t)512 * 512 * 2);
  __bf16* wg_t  = (__bf16*)carve((size_t)512 * 512 * 2);
  __bf16* w1_t  = (__bf16*)carve((size_t)4 * 512 * 1024 * 2);
  __bf16* w2_t  = (__bf16*)carve((size_t)4 * 1024 * 512 * 2);

  const int totalD = BT * D, totalH = BT * Hd;
  const int ebD = (totalD + 255) / 256, ebH = (totalH + 255) / 256;

  // ---- weight transposes (bf16, [N,K]) ----
  wtrans(mp_w, mp_t, 512, 512, stream);
  wtrans(fu_w, fuT_t, 512, 512, stream);
  wtrans(fu_w + (size_t)512 * 512, fuB_t, 512, 512, stream);
  wtrans(wc, wc_t, 512, 32, stream);
  wtrans(wk, wk_t, 32, 512, stream);
  wtrans(wv, wv_t, 32, 512, stream);
  wtrans(wq, wq_t, 512, 512, stream);
  wtrans(wo, wo_t, 512, 512, stream);
  wtrans(wg, wg_t, 512, 512, stream);
  for (int e = 0; e < 4; ++e) {
    wtrans(w1 + (size_t)e * 512 * 1024, w1_t + (size_t)e * 1024 * 512, 512, 1024, stream);
    wtrans(w2 + (size_t)e * 1024 * 512, w2_t + (size_t)e * 512 * 1024, 1024, 512, stream);
  }

  // ---- 1) temporal gating ----
  blk_mean_kernel<<<Bb * NBLK, 256, 0, stream>>>(x, bm);
  tg_gate_kernel<<<Bb, 32, 0, stream>>>(bm, tg_w, tg_b, imp);
  tg_scale_kernel<<<ebD, 256, 0, stream>>>(x, imp, f1, g1, totalD);     // xf32 + xf_bf

  // ---- 2) dual-scale fusion ----
  pool15_kernel<<<ebD, 256, 0, stream>>>(f1, g2, totalD);               // macro (bf16)
  gemm(g2, mp_t, mp_b, nullptr, g3, BT, D, D, 0, 1, stream);            // macro_feat
  gemm(g1, fuT_t, fu_b, f2, nullptr, BT, D, D, 0, 0, stream);           // xfu f32 part 1
  gemm(g3, fuB_t, nullptr, f2, g4, BT, D, D, 1, 1, stream);             // += part 2 -> bf16

  // ---- 3) MLA attention ----
  gemm(g4, wc_t, nullptr, nullptr, gl, BT, 32, D, 0, 1, stream);        // latent
  gemm(gl, wk_t, nullptr, nullptr, g2, BT, D, 32, 0, 1, stream);        // K
  gemm(gl, wv_t, nullptr, nullptr, g3, BT, D, 32, 0, 1, stream);        // V
  gemm(g4, wq_t, nullptr, nullptr, g1, BT, D, D, 0, 1, stream);         // Q
  {
    dim3 g(T / 64, NH, Bb);
    mla_flash_attn_kernel<<<g, 128, 0, stream>>>(g1, g2, g3, g4);       // attnout bf16
  }
  gemm(g4, wo_t, nullptr, f3, nullptr, BT, D, D, 0, 0, stream);         // attn proj f32
  resid_rms_kernel<<<BT, 256, 0, stream>>>(x, f3, s1, f4, g5);          // x1 (f32+bf16)

  // ---- 4) MoE ----
  router_kernel<<<(BT + 255) / 256, 256, 0, stream>>>(f4, wr, br, rw, BT);
  gemm(g5, wg_t, bg, f3, nullptr, BT, D, D, 0, 0, stream);              // gate pre-act
  sigmul_kernel<<<ebD, 256, 0, stream>>>(f4, f3, g5, totalD);           // xg bf16
  for (int e = 0; e < 4; ++e) {
    gemm(g5, w1_t + (size_t)e * 1024 * 512, b1 + (size_t)e * 1024,
         fh, nullptr, BT, Hd, D, 0, 0, stream);                         // h f32
    gelu_kernel<<<ebH, 256, 0, stream>>>(fh, gh, totalH);               // h bf16
    gemm(gh, w2_t + (size_t)e * 512 * 1024, b2 + (size_t)e * 512,
         f2, nullptr, BT, D, Hd, 0, 0, stream);                         // eo f32
    moe_comb_kernel<<<ebD, 256, 0, stream>>>(f1, f2, rw, e, e == 0, totalD);
  }
  resid_rms_kernel<<<BT, 256, 0, stream>>>(f4, f1, s2, out, nullptr);   // final f32
}
